// GaussianAttention_154618823096
// MI455X (gfx1250) — matrix operand order repaired and checked by reading, not compiled, not run on hardware
//
#include <hip/hip_runtime.h>
#include <hip/hip_bf16.h>
#include <cstdint>

// ---------------------------------------------------------------------------
// Problem constants (from reference)
// ---------------------------------------------------------------------------
constexpr long long Bq   = 2;      // batch
constexpr long long Nseq = 2048;   // sequence length
constexpr long long Dm   = 1024;   // d_model
constexpr long long Hh   = 4;      // heads
constexpr long long Dqk  = 256;    // head_qk
constexpr long long Dv   = 512;    // head_v
constexpr long long VD   = 2048;   // value dim
#define EPS_F       1e-5f
#define ROPE_BASE_F 10000.0f
#define QK_SCALE    0.0625f        // 256^-0.5

// ---------------------------------------------------------------------------
// Types / helpers
// ---------------------------------------------------------------------------
typedef __attribute__((ext_vector_type(16))) __bf16 v16bf;
typedef __attribute__((ext_vector_type(8)))  float  v8f;
typedef __attribute__((ext_vector_type(4)))  unsigned int u32x4;
typedef __attribute__((ext_vector_type(8)))  unsigned int u32x8;

__device__ __forceinline__ float bf2f(unsigned short u) {
    return __uint_as_float(((unsigned int)u) << 16);
}
__device__ __forceinline__ unsigned short f2bf(float f) {
    unsigned int u = __float_as_uint(f);
    return (unsigned short)((u + 0x7FFFu + ((u >> 16) & 1u)) >> 16);  // RNE
}
__device__ __forceinline__ __bf16 u2b(unsigned short u) {
    union { unsigned short u; __bf16 b; } x; x.u = u; return x.b;
}

// ---------------------------------------------------------------------------
// f32 -> bf16 conversion kernel (grid-stride)
// ---------------------------------------------------------------------------
__global__ void cvt_f32_bf16(const float* __restrict__ in,
                             unsigned short* __restrict__ out, long long count) {
    long long i = (long long)blockIdx.x * blockDim.x + threadIdx.x;
    long long stride = (long long)gridDim.x * blockDim.x;
    for (; i < count; i += stride) out[i] = f2bf(in[i]);
}

// ---------------------------------------------------------------------------
// Generic strided, batched WMMA GEMM:  acc(m,n) = sum_k A(m,k)*B(k,n)
//   A element (m,k) at  A + zOuter*aOuter + zInner*aInner + m*saM + k*saK (bf16)
//   B element (k,n) at  B + zOuter*bOuter + zInner*bInner + k*sbK + n*sbN (bf16)
//   C element (m,n) at  C + zOuter*cOuter + zInner*cInner + m*ldc + n
// grid: (N/128, M/64, nbatch), block: 128 threads (4 waves, wave32)
// Wave tile 32x64 = 2x4 WMMA fragments -> 8 v_wmma per K-step per wave.
//
// LDS tiles are stored PRE-SWIZZLED in WMMA fragment order:
//   A element (m,k)  -> lAf[m>>4][(m&15) + 16*(k>>4)     ][ k&15              ]
//   B element (k,n)  -> lBf[n>>4][(n&15) + 16*(k>>4)     ][ k&15              ]
// so each lane reads a whole v16bf fragment as one contiguous aligned 32B load.
//
// Triangular structure exploitation:
//   ep==2 (tril-exp epilogue): tiles strictly above the diagonal skip the
//     K-loop entirely and store zeros (uniform branch).
//   aLowerTri: A is lower-triangular -> K-loop clamps to min(K, m0+64).
//
// Epilogues: 0 = store f32, 1 = store bf16,
//            2 = store bf16( row>=col ? exp(scale*acc) (+eps on diag) : 0 )
//            3 = f32  C -= acc   (triangular-solve update)
// ---------------------------------------------------------------------------
struct GemmParams {
    const unsigned short* A;
    const unsigned short* B;
    void* C;
    long long aOuter, aInner, bOuter, bInner, cOuter, cInner;
    long long saM, saK, sbK, sbN, ldc;
    int K;
    int heads;     // inner batch factor (z = zOuter*heads + zInner)
    int ep;
    float scale;
    float eps;
    int addEps;
    int aLowerTri; // clamp K to m0+TILE_M (A lower-triangular)
};

#define TILE_M 64
#define TILE_N 128
#define TILE_K 32

__global__ __launch_bounds__(128) void wmma_gemm(GemmParams p) {
    // [tile16][lane][elem]; elem stride 32 keeps each lane's fragment 32B-aligned
    __shared__ __align__(64) __bf16 lAf[4][32][32];   //  64 x 32 A tile
    __shared__ __align__(64) __bf16 lBf[8][32][32];   //  32 x 128 B tile

    const int tid  = threadIdx.x;
    const int lane = tid & 31;
    const int wave = tid >> 5;
    const int wm   = (wave >> 1) * 32;   // wave M origin (0 / 32)
    const int wn   = (wave & 1) * 64;    // wave N origin (0 / 64)
    const int wmt  = wm >> 4;            // A tile16 base index
    const int wnt  = wn >> 4;            // B tile16 base index

    const int z  = blockIdx.z;
    const int zo = z / p.heads;
    const int zi = z % p.heads;
    const unsigned short* Ab = p.A + (long long)zo * p.aOuter + (long long)zi * p.aInner;
    const unsigned short* Bb = p.B + (long long)zo * p.bOuter + (long long)zi * p.bInner;
    const long long cbase = (long long)zo * p.cOuter + (long long)zi * p.cInner;

    const int m0 = blockIdx.y * TILE_M;
    const int n0 = blockIdx.x * TILE_N;

    const int mr = lane & 15;            // C frag column / A frag row
    const int mo = (lane < 16) ? 0 : 8;  // C frag M offset per ISA layout

    // ---- tril tile skip: strictly-upper tiles of exp(..) epilogue are zero ----
    if (p.ep == 2 && n0 > m0 + (TILE_M - 1)) {
        #pragma unroll
        for (int tm = 0; tm < 2; ++tm)
            #pragma unroll
            for (int tn = 0; tn < 4; ++tn)
                #pragma unroll
                for (int e = 0; e < 8; ++e) {
                    int gm = m0 + wm + tm * 16 + mo + e;
                    int gn = n0 + wn + tn * 16 + mr;
                    ((unsigned short*)p.C)[cbase + (long long)gm * p.ldc + gn] = 0;
                }
        return;
    }

    int Keff = p.K;
    if (p.aLowerTri) {                  // A cols beyond block-row diagonal are zero
        int kl = m0 + TILE_M;
        Keff = (Keff < kl) ? Keff : kl;
    }

    v8f acc[2][4] = {};                 // 8x fp32 16x16 accumulators per wave

    for (int kb = 0; kb < Keff; kb += TILE_K) {
        // ---------------- stage A tile (64 x 32) ----------------
        if (p.saK == 1) {
            for (int c = tid; c < 256; c += 128) {     // 16B chunks along k
                int m = c >> 2, kc = (c & 3) * 8;
                uint4 v = *(const uint4*)(Ab + (long long)(m0 + m) * p.saM + (kb + kc));
                *(uint4*)&lAf[m >> 4][(m & 15) + ((kc >> 4) & 1) * 16][kc & 8] = v;
            }
        } else {
            for (int idx = tid; idx < TILE_M * TILE_K; idx += 128) {
                int m = idx >> 5, k = idx & 31;
                lAf[m >> 4][(m & 15) + ((k >> 4) & 1) * 16][k & 15] =
                    u2b(Ab[(long long)(m0 + m) * p.saM + (long long)(kb + k) * p.saK]);
            }
        }
        // ---------------- stage B tile (32 x 128) ----------------
        if (p.sbK == 1) {
            for (int c = tid; c < 512; c += 128) {     // 16B chunks along k
                int n = c >> 2, kc = (c & 3) * 8;
                uint4 v = *(const uint4*)(Bb + (long long)(n0 + n) * p.sbN + (kb + kc));
                *(uint4*)&lBf[n >> 4][(n & 15) + ((kc >> 4) & 1) * 16][kc & 8] = v;
            }
        } else if (p.sbN == 1) {
            for (int c = tid; c < 512; c += 128) {     // 16B load, 8x u16 scatter
                int k = c >> 4, nc = (c & 15) * 8;
                uint4 v = *(const uint4*)(Bb + (long long)(kb + k) * p.sbK + (n0 + nc));
                const unsigned short* pv = (const unsigned short*)&v;
                #pragma unroll
                for (int j = 0; j < 8; ++j) {
                    int n = nc + j;
                    lBf[n >> 4][(n & 15) + ((k >> 4) & 1) * 16][k & 15] = u2b(pv[j]);
                }
            }
        } else {
            for (int idx = tid; idx < TILE_K * TILE_N; idx += 128) {
                int k = idx >> 7, n = idx & 127;
                lBf[n >> 4][(n & 15) + ((k >> 4) & 1) * 16][k & 15] =
                    u2b(Bb[(long long)(kb + k) * p.sbK + (long long)(n0 + n) * p.sbN]);
            }
        }
        if (kb + TILE_K < Keff) {  // hint next K-slab toward L2 (global_prefetch_b8)
            __builtin_prefetch(Ab + (long long)(m0 + (tid >> 1)) * p.saM +
                                   (long long)(kb + TILE_K + (tid & 1) * 16) * p.saK, 0, 1);
        }
        __syncthreads();

        // ---------------- fragments: one aligned 32B LDS read each ----------
        v16bf af[2], bfr[4];
        #pragma unroll
        for (int t = 0; t < 2; ++t) af[t] = *(const v16bf*)&lAf[wmt + t][lane][0];
        #pragma unroll
        for (int t = 0; t < 4; ++t) bfr[t] = *(const v16bf*)&lBf[wnt + t][lane][0];

        #pragma unroll
        for (int tm = 0; tm < 2; ++tm)
            #pragma unroll
            for (int tn = 0; tn < 4; ++tn)
                acc[tm][tn] = __builtin_amdgcn_wmma_f32_16x16x32_bf16(
                    false, af[tm], false, bfr[tn], (short)0, acc[tm][tn], false, false);
        __syncthreads();
    }

    // ---- epilogue ----
    #pragma unroll
    for (int tm = 0; tm < 2; ++tm) {
        #pragma unroll
        for (int tn = 0; tn < 4; ++tn) {
            #pragma unroll
            for (int e = 0; e < 8; ++e) {
                int gm = m0 + wm + tm * 16 + mo + e;
                int gn = n0 + wn + tn * 16 + mr;
                float v = acc[tm][tn][e];
                long long cix = cbase + (long long)gm * p.ldc + gn;
                if (p.ep == 0) {
                    ((float*)p.C)[cix] = v;
                } else if (p.ep == 1) {
                    ((unsigned short*)p.C)[cix] = f2bf(v);
                } else if (p.ep == 2) {
                    float r = 0.0f;
                    if (gm >= gn) {
                        r = __expf(v * p.scale);
                        if (p.addEps && gm == gn) r += p.eps;
                    }
                    ((unsigned short*)p.C)[cix] = f2bf(r);
                } else {  // ep == 3 : C -= acc (fp32 RMW, solve update)
                    ((float*)p.C)[cix] -= v;
                }
            }
        }
    }
}

// ---------------------------------------------------------------------------
// RoPE + head transpose:  in (b,n,H,hd) bf16  ->  out (b,H,n,hd) bf16
// ---------------------------------------------------------------------------
__global__ void rope_transpose(const unsigned short* __restrict__ in,
                               unsigned short* __restrict__ out) {
    const int hd = (int)Dqk, half = hd >> 1;
    long long idx   = (long long)blockIdx.x * blockDim.x + threadIdx.x;
    long long total = Bq * Nseq * Hh * half;
    if (idx >= total) return;
    int j = (int)(idx % half); long long t = idx / half;
    int h = (int)(t % Hh);     t /= Hh;
    int pos = (int)(t % Nseq); int b = (int)(t / Nseq);

    const unsigned short* src = in  + (((long long)b * Nseq + pos) * Hh + h) * hd;
    unsigned short*       dst = out + (((long long)b * Hh + h) * Nseq + pos) * hd;

    float t1 = bf2f(src[j]);
    float t2 = bf2f(src[j + half]);
    float freq = __powf(ROPE_BASE_F, -((float)(2 * j)) / (float)hd);
    float ang  = (float)pos * freq;
    float c = __cosf(ang), s = __sinf(ang);
    dst[j]        = f2bf(t1 * c - t2 * s);
    dst[j + half] = f2bf(t2 * c + t1 * s);
}

// ---------------------------------------------------------------------------
// Split V heads:  in (b,n,H*Dv) f32  ->  out (b,H,n,Dv) f32 (initial RHS)
// ---------------------------------------------------------------------------
__global__ void split_heads_v(const float* __restrict__ in, float* __restrict__ out) {
    long long idx   = (long long)blockIdx.x * blockDim.x + threadIdx.x;
    long long total = Bq * Nseq * Hh * Dv;
    if (idx >= total) return;
    int d = (int)(idx % Dv); long long t = idx / Dv;
    int h = (int)(t % Hh);   t /= Hh;
    int pos = (int)(t % Nseq); int b = (int)(t / Nseq);
    out[(((long long)b * Hh + h) * Nseq + pos) * Dv + d] =
        in[((long long)b * Nseq + pos) * VD + (long long)h * Dv + d];
}

// ---------------------------------------------------------------------------
// Diagonal-block forward substitution for the blocked triangular solve.
// The 64x64 diagonal block of exp_kk is staged into LDS by the Tensor Data
// Mover (tensor_load_to_lds, descriptor per ISA cdna5 ch.8), completed with
// s_wait_tensorcnt. One block per (b,h); one thread per RHS column.
// ---------------------------------------------------------------------------
__global__ __launch_bounds__(512) void trisolve_diag(const unsigned short* __restrict__ Akk,
                                                     float* __restrict__ sol,
                                                     unsigned short* __restrict__ solb,
                                                     int iblk) {
    const int bh  = blockIdx.y;
    const int col = threadIdx.x;  // 0..511
    __shared__ __align__(16) unsigned short Araw[64 * 64];
    __shared__ float As[64][65];

    const unsigned short* Ab = Akk + (long long)bh * Nseq * Nseq
                                   + (long long)iblk * 64 * Nseq + (long long)iblk * 64;

    if (threadIdx.x < 32) {  // wave 0 issues the TDM DMA (EXEC ignored by TENSOR ops)
        unsigned long long ga = (unsigned long long)(uintptr_t)Ab;       // global byte addr
        unsigned int ldsa     = (unsigned int)(uintptr_t)&Araw[0];       // LDS byte offset

        u32x4 g0;                       // D# group 0 (ISA 8.3)
        g0[0] = 1u;                                             // count=1, user mode
        g0[1] = ldsa;                                           // lds_addr
        g0[2] = (unsigned int)(ga & 0xffffffffu);               // global_addr[31:0]
        g0[3] = (unsigned int)((ga >> 32) & 0x01ffffffu)        // global_addr[56:32]
              | (2u << 30);                                     // type=2 ("image")

        u32x8 g1;                       // D# group 1 (ISA 8.4)
        g1[0] = 0x00010000u;            // workgroup_mask=0, data_size=1 (2 bytes)
        g1[1] = 0x00000000u;            // atomic barrier addr=0, tensor_dim0[15:0]=0
        g1[2] = 0x00001000u;            // tensor_dim0[31:16]=0x1000, tensor_dim1[15:0]=0
        g1[3] = (64u << 16) | 0x1000u;  // tensor_dim1[31:16]=0x1000, tile_dim0=64
        g1[4] = 64u;                    // tile_dim1=64, tile_dim2=0
        g1[5] = 2048u;                  // tensor_dim0_stride[31:0] = Nseq
        g1[6] = 0u;                     // stride0[47:32]=0, stride1[15:0]=0
        g1[7] = 0u;

        asm volatile("tensor_load_to_lds %0, %1" :: "s"(g0), "s"(g1) : "memory");
        __builtin_amdgcn_s_wait_tensorcnt(0);
    }
    __syncthreads();

    for (int idx = threadIdx.x; idx < 64 * 64; idx += 512) {
        int r = idx >> 6, c = idx & 63;
        As[r][c] = bf2f(Araw[idx]);
    }
    __syncthreads();

    long long base = ((long long)bh * Nseq + (long long)iblk * 64) * Dv + col;
    float xv[64];
    for (int r = 0; r < 64; ++r) {
        float a = sol[base + (long long)r * Dv];
        for (int c = 0; c < r; ++c) a -= As[r][c] * xv[c];
        float x = a / As[r][r];
        xv[r] = x;
        sol[base + (long long)r * Dv]  = x;
        solb[base + (long long)r * Dv] = f2bf(x);
    }
}

// ---------------------------------------------------------------------------
// Host side
// ---------------------------------------------------------------------------
static inline void launch_gemm(hipStream_t stream, const GemmParams& p, int M, int Nn, int nbatch) {
    dim3 grid((unsigned)(Nn / TILE_N), (unsigned)(M / TILE_M), (unsigned)nbatch);
    wmma_gemm<<<grid, dim3(128), 0, stream>>>(p);
}

extern "C" void kernel_launch(void* const* d_in, const int* in_sizes, int n_in,
                              void* d_out, int out_size, void* d_ws, size_t ws_size,
                              hipStream_t stream) {
    (void)in_sizes; (void)n_in; (void)out_size; (void)ws_size;

    const float* x  = (const float*)d_in[0];
    const float* Wq = (const float*)d_in[1];
    const float* Wk = (const float*)d_in[2];
    const float* Wv = (const float*)d_in[3];
    const float* Wo = (const float*)d_in[4];
    float* out = (float*)d_out;

    // ---- workspace carve-up ----
    char* ws = (char*)d_ws;
    size_t off = 0;
    auto carve = [&](size_t bytes) { char* p = ws + off; off += (bytes + 255) & ~(size_t)255; return p; };
    unsigned short* xb    = (unsigned short*)carve(Bq * Nseq * Dm * 2);
    unsigned short* wqb   = (unsigned short*)carve(Dm * 1024 * 2);
    unsigned short* wkb   = (unsigned short*)carve(Dm * 1024 * 2);
    unsigned short* wvb   = (unsigned short*)carve(Dm * VD * 2);
    unsigned short* wob   = (unsigned short*)carve(VD * Dm * 2);
    unsigned short* qtmp  = (unsigned short*)carve(Bq * Nseq * 1024 * 2);
    unsigned short* ktmp  = (unsigned short*)carve(Bq * Nseq * 1024 * 2);
    float*          vtmp  = (float*)         carve(Bq * Nseq * VD * 4);
    unsigned short* qr    = (unsigned short*)carve(Bq * Nseq * 1024 * 2);
    unsigned short* kr    = (unsigned short*)carve(Bq * Nseq * 1024 * 2);
    unsigned short* eqk   = (unsigned short*)carve(Bq * Hh * Nseq * Nseq * 2);
    unsigned short* ekk   = (unsigned short*)carve(Bq * Hh * Nseq * Nseq * 2);
    float*          solf  = (float*)         carve(Bq * Hh * Nseq * Dv * 4);
    unsigned short* solb  = (unsigned short*)carve(Bq * Hh * Nseq * Dv * 2);
    unsigned short* obf   = (unsigned short*)carve(Bq * Nseq * VD * 2);

    // ---- 1) convert inputs to bf16 ----
    cvt_f32_bf16<<<4096, 256, 0, stream>>>(x,  xb,  Bq * Nseq * Dm);
    cvt_f32_bf16<<<1024, 256, 0, stream>>>(Wq, wqb, Dm * 1024);
    cvt_f32_bf16<<<1024, 256, 0, stream>>>(Wk, wkb, Dm * 1024);
    cvt_f32_bf16<<<2048, 256, 0, stream>>>(Wv, wvb, Dm * VD);
    cvt_f32_bf16<<<2048, 256, 0, stream>>>(Wo, wob, VD * Dm);

    // ---- 2) projections ----
    GemmParams p = {};
    p.A = xb;  p.aOuter = Nseq * Dm; p.aInner = 0; p.saM = Dm; p.saK = 1;
    p.B = wqb; p.bOuter = 0;         p.bInner = 0; p.sbK = 1024; p.sbN = 1;
    p.C = qtmp; p.cOuter = Nseq * 1024; p.cInner = 0; p.ldc = 1024;
    p.K = (int)Dm; p.heads = 1; p.ep = 1; p.scale = 1.f; p.eps = 0.f; p.addEps = 0; p.aLowerTri = 0;
    launch_gemm(stream, p, (int)Nseq, 1024, (int)Bq);
    p.B = wkb; p.C = ktmp;
    launch_gemm(stream, p, (int)Nseq, 1024, (int)Bq);
    p.B = wvb; p.sbK = VD; p.sbN = 1;
    p.C = vtmp; p.cOuter = Nseq * VD; p.ldc = VD; p.ep = 0;
    launch_gemm(stream, p, (int)Nseq, (int)VD, (int)Bq);

    // ---- 3) RoPE + head transpose; split V into (b,h,n,dv) initial RHS ----
    {
        long long tot = Bq * Nseq * Hh * (Dqk / 2);
        rope_transpose<<<(unsigned)((tot + 255) / 256), 256, 0, stream>>>(qtmp, qr);
        rope_transpose<<<(unsigned)((tot + 255) / 256), 256, 0, stream>>>(ktmp, kr);
        long long tv = Bq * Nseq * Hh * Dv;
        split_heads_v<<<(unsigned)((tv + 255) / 256), 256, 0, stream>>>(vtmp, solf);
    }

    // ---- 4) exp_qk = tril(exp(q k^T /16)); exp_kk likewise + eps*I ----
    // (strictly-upper 64x128 tiles skip their K-loop and store zeros)
    GemmParams g = {};
    g.A = qr; g.aOuter = Hh * Nseq * Dqk; g.aInner = Nseq * Dqk; g.saM = Dqk; g.saK = 1;
    g.B = kr; g.bOuter = Hh * Nseq * Dqk; g.bInner = Nseq * Dqk; g.sbK = 1;   g.sbN = Dqk;
    g.C = eqk; g.cOuter = Hh * Nseq * Nseq; g.cInner = Nseq * Nseq; g.ldc = Nseq;
    g.K = (int)Dqk; g.heads = (int)Hh; g.ep = 2; g.scale = QK_SCALE; g.eps = EPS_F;
    g.addEps = 0; g.aLowerTri = 0;
    launch_gemm(stream, g, (int)Nseq, (int)Nseq, (int)(Bq * Hh));
    g.A = kr; g.C = ekk; g.addEps = 1;
    launch_gemm(stream, g, (int)Nseq, (int)Nseq, (int)(Bq * Hh));

    // ---- 5) blocked lower-triangular solve: A sol = V, NB = 64 ----
    for (int i = 0; i < (int)(Nseq / 64); ++i) {
        if (i > 0) {
            GemmParams u = {};
            u.A = ekk + (long long)i * 64 * Nseq;
            u.aOuter = Hh * Nseq * Nseq; u.aInner = Nseq * Nseq; u.saM = Nseq; u.saK = 1;
            u.B = solb; u.bOuter = Hh * Nseq * Dv; u.bInner = Nseq * Dv; u.sbK = Dv; u.sbN = 1;
            u.C = solf + (long long)i * 64 * Dv;
            u.cOuter = Hh * Nseq * Dv; u.cInner = Nseq * Dv; u.ldc = Dv;
            u.K = i * 64; u.heads = (int)Hh; u.ep = 3; u.scale = 1.f; u.eps = 0.f;
            u.addEps = 0; u.aLowerTri = 0;
            launch_gemm(stream, u, 64, (int)Dv, (int)(Bq * Hh));
        }
        trisolve_diag<<<dim3(1, (unsigned)(Bq * Hh)), 512, 0, stream>>>(ekk, solf, solb, i);
    }

    // ---- 6) o = exp_qk @ sol  (A lower-triangular -> K clamped per block row) ----
    GemmParams o = {};
    o.A = eqk; o.aOuter = Hh * Nseq * Nseq; o.aInner = Nseq * Nseq; o.saM = Nseq; o.saK = 1;
    o.B = solb; o.bOuter = Hh * Nseq * Dv; o.bInner = Nseq * Dv; o.sbK = Dv; o.sbN = 1;
    o.C = obf; o.cOuter = Nseq * VD; o.cInner = Dv; o.ldc = VD;
    o.K = (int)Nseq; o.heads = (int)Hh; o.ep = 1; o.scale = 1.f; o.eps = 0.f;
    o.addEps = 0; o.aLowerTri = 1;
    launch_gemm(stream, o, (int)Nseq, (int)Dv, (int)(Bq * Hh));

    // ---- 7) out = o @ Wo  -> f32 (b,n,1024) ----
    GemmParams f = {};
    f.A = obf; f.aOuter = Nseq * VD; f.aInner = 0; f.saM = VD; f.saK = 1;
    f.B = wob; f.bOuter = 0; f.bInner = 0; f.sbK = 1024; f.sbN = 1;
    f.C = out; f.cOuter = Nseq * 1024; f.cInner = 0; f.ldc = 1024;
    f.K = (int)VD; f.heads = 1; f.ep = 0; f.scale = 1.f; f.eps = 0.f;
    f.addEps = 0; f.aLowerTri = 0;
    launch_gemm(stream, f, (int)Nseq, 1024, (int)Bq);
}
